// PPAggregator_65214783422876
// MI455X (gfx1250) — compile-verified, matrix operand order checked by reading, not run on hardware
//
#include <hip/hip_runtime.h>

typedef __attribute__((ext_vector_type(16))) _Float16 v16h;
typedef __attribute__((ext_vector_type(8)))  float    v8f;

#define DD   128
#define KN   32
#define XS   264   // Xh row stride (halfs): 2*D + 8 pad -> conflict-free A-frag gathers
#define H1S  136   // H1 row stride (halfs): D + 8 pad
#define H2S  129   // H2 row stride (floats): D + 1 pad
#define NEGV (-1e9f)

union AFrag { v16h v; unsigned u[8]; };

// ---------------------------------------------------------------------------
// Prep: swizzle W1 [256x128] f32 and W2 [128x128] f32 into per-lane WMMA
// B-fragment layout (f16) in workspace. Fragment (kt, nt): lane L holds
// column n = nt*16 + L%16, elements i: k = kt*32 + (i/8)*16 + (L/16)*8 + i%8.
// ---------------------------------------------------------------------------
__global__ __launch_bounds__(256) void prep_weights_kernel(
    const float* __restrict__ W1, const float* __restrict__ W2,
    _Float16* __restrict__ w1h, _Float16* __restrict__ w2h) {
  int idx = blockIdx.x * blockDim.x + threadIdx.x;
  const int total1 = 8 * 8 * 32 * 16;  // 32768 (kt=8, nt=8)
  const int total2 = 4 * 8 * 32 * 16;  // 16384 (kt=4, nt=8)
  if (idx < total1) {
    int i = idx & 15, lane = (idx >> 4) & 31, nt = (idx >> 9) & 7, kt = idx >> 12;
    int k = kt * 32 + ((i >> 3) << 4) + ((lane >> 4) << 3) + (i & 7);
    int n = (nt << 4) + (lane & 15);
    w1h[idx] = (_Float16)W1[k * DD + n];
  } else if (idx < total1 + total2) {
    int j = idx - total1;
    int i = j & 15, lane = (j >> 4) & 31, nt = (j >> 9) & 7, kt = j >> 12;
    int k = kt * 32 + ((i >> 3) << 4) + ((lane >> 4) << 3) + (i & 7);
    int n = (nt << 4) + (lane & 15);
    w2h[j] = (_Float16)W2[k * DD + n];
  }
}

// ---------------------------------------------------------------------------
// Main: one block per node. 8 waves; wave w owns N-tile w (16 columns).
// ---------------------------------------------------------------------------
__global__ __launch_bounds__(256) void ppagg_kernel(
    const int* __restrict__ nodes, const int* __restrict__ neigh,
    const int* __restrict__ degs,  const float* __restrict__ u2e,
    const float* __restrict__ b1,  const float* __restrict__ b2,
    const float* __restrict__ W3,  const float* __restrict__ b3,
    const _Float16* __restrict__ w1h, const _Float16* __restrict__ w2h,
    float* __restrict__ out) {
  __shared__ _Float16 Xh[KN * XS];     // [32 x 256] f16 (padded)
  __shared__ float    Eu[KN * DD];     // neighbor feats f32 (for weighted sum)
  __shared__ _Float16 H1[KN * H1S];    // layer-1 output f16 (padded)
  __shared__ float    H2[KN * H2S];    // layer-2 output f32 (padded)
  __shared__ float    Sf[DD];
  __shared__ float    Att[KN];

  const int n    = blockIdx.x;
  const int tid  = threadIdx.x;
  const int lane = tid & 31;
  const int wv   = tid >> 5;           // 0..7 = N-tile
  const int deg  = degs[n];
  const int khalf = lane >> 4;
  const int mrow  = lane & 15;

  // ---- self features
  if (tid < DD) Sf[tid] = u2e[(long)nodes[n] * DD + tid];
  // ---- gather neighbors (coalesced per row), keep f32 + f16 copies
  for (int idx = tid; idx < KN * DD; idx += 256) {
    int r = idx >> 7, c = idx & 127;
    float v = u2e[(long)neigh[n * KN + r] * DD + c];
    Eu[r * DD + c]  = v;
    Xh[r * XS + c]  = (_Float16)v;
  }
  __syncthreads();
  // ---- broadcast self into X[:,128:256]
  for (int idx = tid; idx < KN * DD; idx += 256) {
    int r = idx >> 7, c = idx & 127;
    Xh[r * XS + DD + c] = (_Float16)Sf[c];
  }
  __syncthreads();

  const int ncol = (wv << 4) + mrow;

  // ================= Layer 1: X[32x256] @ W1 -> relu -> H1[32x128] ========
  {
    v8f acc0 = {}; v8f acc1 = {};
#pragma unroll
    for (int kt = 0; kt < 8; ++kt) {
      v16h bf = *(const v16h*)(w1h + ((kt * 8 + wv) * 512 + lane * 16));
      AFrag a0, a1;
#pragma unroll
      for (int j = 0; j < 8; ++j) {
        int kp = kt * 32 + ((j >> 2) << 4) + (khalf << 3) + ((j & 3) << 1);
        a0.u[j] = *(const unsigned*)(&Xh[mrow * XS + kp]);
        a1.u[j] = *(const unsigned*)(&Xh[(16 + mrow) * XS + kp]);
      }
      acc0 = __builtin_amdgcn_wmma_f32_16x16x32_f16(false, a0.v, false, bf,
                                                    (short)0, acc0, false, false);
      acc1 = __builtin_amdgcn_wmma_f32_16x16x32_f16(false, a1.v, false, bf,
                                                    (short)0, acc1, false, false);
    }
    float bias = b1[ncol];
#pragma unroll
    for (int r = 0; r < 8; ++r) {
      int m = (khalf << 3) + r;   // C/D layout: VGPR r -> M = r + 8*(lane/16)
      H1[m * H1S + ncol]        = (_Float16)fmaxf(acc0[r] + bias, 0.f);
      H1[(16 + m) * H1S + ncol] = (_Float16)fmaxf(acc1[r] + bias, 0.f);
    }
  }
  __syncthreads();

  // ================= Layer 2: H1[32x128] @ W2 -> relu -> H2[32x128] =======
  {
    v8f acc0 = {}; v8f acc1 = {};
#pragma unroll
    for (int kt = 0; kt < 4; ++kt) {
      v16h bf = *(const v16h*)(w2h + ((kt * 8 + wv) * 512 + lane * 16));
      AFrag a0, a1;
#pragma unroll
      for (int j = 0; j < 8; ++j) {
        int kp = kt * 32 + ((j >> 2) << 4) + (khalf << 3) + ((j & 3) << 1);
        a0.u[j] = *(const unsigned*)(&H1[mrow * H1S + kp]);
        a1.u[j] = *(const unsigned*)(&H1[(16 + mrow) * H1S + kp]);
      }
      acc0 = __builtin_amdgcn_wmma_f32_16x16x32_f16(false, a0.v, false, bf,
                                                    (short)0, acc0, false, false);
      acc1 = __builtin_amdgcn_wmma_f32_16x16x32_f16(false, a1.v, false, bf,
                                                    (short)0, acc1, false, false);
    }
    float bias = b2[ncol];
#pragma unroll
    for (int r = 0; r < 8; ++r) {
      int m = (khalf << 3) + r;
      H2[m * H2S + ncol]        = fmaxf(acc0[r] + bias, 0.f);
      H2[(16 + m) * H2S + ncol] = fmaxf(acc1[r] + bias, 0.f);
    }
  }
  __syncthreads();

  // ============ Layer 3 + masked softmax (wave 0, lane = neighbor) ========
  if (wv == 0) {
    float acc = b3[0];
#pragma unroll 4
    for (int d = 0; d < DD; ++d) acc += H2[lane * H2S + d] * W3[d];
    float lg = (lane < deg) ? acc : NEGV;
    float mx = lg;
    for (int off = 16; off > 0; off >>= 1) mx = fmaxf(mx, __shfl_xor(mx, off, 32));
    float e = __expf(lg - mx);
    float s = e;
    for (int off = 16; off > 0; off >>= 1) s += __shfl_xor(s, off, 32);
    Att[lane] = e / s;
  }
  __syncthreads();

  // ============ Attention-weighted aggregation (f32) ======================
  if (tid < DD) {
    float acc = 0.f;
#pragma unroll 4
    for (int k = 0; k < KN; ++k) acc += Att[k] * Eu[k * DD + tid];
    float self = Sf[tid];
    out[(long)n * DD + tid] = (deg > 0) ? 0.5f * (acc + self) : self;
  }
}

// ---------------------------------------------------------------------------
extern "C" void kernel_launch(void* const* d_in, const int* in_sizes, int n_in,
                              void* d_out, int out_size, void* d_ws, size_t ws_size,
                              hipStream_t stream) {
  const int*   nodes = (const int*)d_in[0];
  const int*   neigh = (const int*)d_in[1];
  const int*   degs  = (const int*)d_in[2];
  const float* u2e   = (const float*)d_in[3];
  const float* W1    = (const float*)d_in[4];
  const float* b1    = (const float*)d_in[5];
  const float* W2    = (const float*)d_in[6];
  const float* b2    = (const float*)d_in[7];
  const float* W3    = (const float*)d_in[8];
  const float* b3    = (const float*)d_in[9];
  float* out = (float*)d_out;

  _Float16* w1h = (_Float16*)d_ws;        // 32768 halfs = 64 KB
  _Float16* w2h = w1h + 32768;            // 16384 halfs = 32 KB

  const int N = in_sizes[0];              // 20000

  prep_weights_kernel<<<(32768 + 16384 + 255) / 256, 256, 0, stream>>>(W1, W2, w1h, w2h);
  ppagg_kernel<<<N, 256, 0, stream>>>(nodes, neigh, degs, u2e, b1, b2, W3, b3,
                                      w1h, w2h, out);
}
